// BitLinear_12094627905998
// MI455X (gfx1250) — matrix operand order repaired and checked by reading, not compiled, not run on hardware
//
#include <hip/hip_runtime.h>
#include <hip/hip_bf16.h>

typedef __attribute__((ext_vector_type(4)))  __bf16 v4bf;
typedef __attribute__((ext_vector_type(8)))  __bf16 v8bf;
typedef __attribute__((ext_vector_type(16))) __bf16 v16bf;
typedef __attribute__((ext_vector_type(8)))  float  v8f;
typedef int v4i128 __attribute__((__vector_size__(16)));

constexpr int MDIM = 8192;   // B*S = 4*2048
constexpr int NDIM = 4096;   // D_OUT
constexpr int KDIM = 4096;   // D_IN
constexpr long long WELEMS = (long long)NDIM * KDIM;   // 16,777,216
constexpr long long XELEMS = (long long)MDIM * KDIM;   // 33,554,432
constexpr int ABS_BLOCKS = 1024;
constexpr int GEMM_BLOCKS = (MDIM / 128) * (NDIM / 256); // 64*16 = 1024
constexpr int LDS_STRIDE = 40;   // 32 bf16 cols padded to 40 (80B) -> conflict-free b128 reads

// -------- CDNA5 async-to-LDS probe (falls back to sync copy if absent) --------
#if defined(__has_builtin)
#  if __has_builtin(__builtin_amdgcn_global_load_async_to_lds_b128)
#    define USE_ASYNC_LDS 1
#  endif
#endif
#ifndef USE_ASYNC_LDS
#  define USE_ASYNC_LDS 0
#endif

__device__ __forceinline__ void wait_async_lds() {
#if USE_ASYNC_LDS
#  if __has_builtin(__builtin_amdgcn_s_wait_asynccnt)
  __builtin_amdgcn_s_wait_asynccnt(0);
#  else
  asm volatile("s_wait_asynccnt 0x0" ::: "memory");
#  endif
#endif
}

// Copy one 16B chunk global -> LDS (async if available).
__device__ __forceinline__ void copy_chunk(const __bf16* g, __bf16* l) {
#if USE_ASYNC_LDS
  __builtin_amdgcn_global_load_async_to_lds_b128(
      (__attribute__((address_space(1))) v4i128*)g,
      (__attribute__((address_space(3))) v4i128*)l, 0, 0);
#else
  *(uint4*)l = *(const uint4*)g;
#endif
}

// Stage a ROWS x 32 bf16 tile (row-major, global row stride KDIM) into LDS
// (row stride LDS_STRIDE). 256 threads, each moves ROWS*4/256 16B chunks.
template <int ROWS>
__device__ __forceinline__ void stage_tile(const __bf16* __restrict__ g0,
                                           __bf16* l0, int tid) {
#pragma unroll
  for (int i = 0; i < ROWS * 4 / 256; ++i) {
    const int c = tid + i * 256;
    const int row = c >> 2, q = c & 3;
    copy_chunk(g0 + (size_t)row * KDIM + q * 8, l0 + row * LDS_STRIDE + q * 8);
  }
}

// ---------------- 1) per-block partial sums of |weight| ----------------
__global__ __launch_bounds__(256) void abssum_partial(const float4* __restrict__ w,
                                                      float* __restrict__ part) {
  __shared__ float red[256];
  float acc = 0.f;
  for (long long i = (long long)blockIdx.x * 256 + threadIdx.x; i < WELEMS / 4;
       i += (long long)gridDim.x * 256) {
    float4 v = w[i];
    acc += fabsf(v.x) + fabsf(v.y) + fabsf(v.z) + fabsf(v.w);
  }
  red[threadIdx.x] = acc;
  __syncthreads();
  for (int o = 128; o > 0; o >>= 1) {
    if (threadIdx.x < (unsigned)o) red[threadIdx.x] += red[threadIdx.x + o];
    __syncthreads();
  }
  if (threadIdx.x == 0) part[blockIdx.x] = red[0];
}

// ---------------- 2) threshold = 0.7 * |scale| * mean(|w|) ----------------
__global__ __launch_bounds__(256) void finalize_thr(const float* __restrict__ part,
                                                    const float* __restrict__ scale,
                                                    float* __restrict__ thr) {
  __shared__ float red[256];
  float a = 0.f;
  for (int i = threadIdx.x; i < ABS_BLOCKS; i += 256) a += part[i];
  red[threadIdx.x] = a;
  __syncthreads();
  for (int o = 128; o > 0; o >>= 1) {
    if (threadIdx.x < (unsigned)o) red[threadIdx.x] += red[threadIdx.x + o];
    __syncthreads();
  }
  if (threadIdx.x == 0)
    thr[0] = 0.7f * fabsf(scale[0]) * red[0] / (float)WELEMS;
}

// ---------------- 3) ternarize weights -> bf16 ----------------
__global__ __launch_bounds__(256) void quant_w(const float4* __restrict__ w,
                                               const float* __restrict__ scale,
                                               const float* __restrict__ thr,
                                               v4bf* __restrict__ wq) {
  const float s = scale[0], t = thr[0];
  for (long long i = (long long)blockIdx.x * 256 + threadIdx.x; i < WELEMS / 4;
       i += (long long)gridDim.x * 256) {
    float4 v = w[i];
    v4bf q;
    q[0] = (__bf16)((v.x * s > t) ? 1.0f : ((v.x * s < -t) ? -1.0f : 0.0f));
    q[1] = (__bf16)((v.y * s > t) ? 1.0f : ((v.y * s < -t) ? -1.0f : 0.0f));
    q[2] = (__bf16)((v.z * s > t) ? 1.0f : ((v.z * s < -t) ? -1.0f : 0.0f));
    q[3] = (__bf16)((v.w * s > t) ? 1.0f : ((v.w * s < -t) ? -1.0f : 0.0f));
    wq[i] = q;
  }
}

// ---------------- 4) x fp32 -> bf16 ----------------
__global__ __launch_bounds__(256) void cvt_x(const float4* __restrict__ x,
                                             v4bf* __restrict__ xb) {
  for (long long i = (long long)blockIdx.x * 256 + threadIdx.x; i < XELEMS / 4;
       i += (long long)gridDim.x * 256) {
    float4 v = x[i];
    v4bf q;
    q[0] = (__bf16)v.x; q[1] = (__bf16)v.y; q[2] = (__bf16)v.z; q[3] = (__bf16)v.w;
    xb[i] = q;
  }
}

// ---------------- 5) WMMA GEMM: out = Xb * Wq^T + bias, track min/max ---------
// Block tile 128(M) x 256(N), 8 waves (2Mx4N), each wave 64x64 = 4x4 WMMA tiles.
// Double-buffered LDS staging (async-to-LDS when available), K step 32.
__global__ __launch_bounds__(256) void gemm_bitlinear(const __bf16* __restrict__ X,
                                                      const __bf16* __restrict__ W,
                                                      const float* __restrict__ bias,
                                                      float* __restrict__ out,
                                                      float* __restrict__ bmin,
                                                      float* __restrict__ bmax) {
  __shared__ __bf16 As[2][128 * LDS_STRIDE];  // 2 x 10 KiB
  __shared__ __bf16 Bs[2][256 * LDS_STRIDE];  // 2 x 20 KiB

  const int tid  = threadIdx.x;
  const int lane = tid & 31;
  const int wid  = tid >> 5;   // 0..7
  const int wm   = wid & 1;    // 2 waves along M
  const int wn   = wid >> 1;   // 4 waves along N
  const int r    = lane & 15;
  const int h    = lane >> 4;

  const int m0 = blockIdx.y * 128 + wm * 64;
  const int n0 = blockIdx.x * 256 + wn * 64;

  const __bf16* Ablk = X + (size_t)(blockIdx.y * 128) * KDIM;
  const __bf16* Bblk = W + (size_t)(blockIdx.x * 256) * KDIM;

  v8f acc[4][4] = {};

  constexpr int KT = KDIM / 32;  // 128 steps
  // Prologue: stage tile 0 into buffer 0
  stage_tile<128>(Ablk, &As[0][0], tid);
  stage_tile<256>(Bblk, &Bs[0][0], tid);

  for (int kt = 0; kt < KT; ++kt) {
    const int cur = kt & 1;
    wait_async_lds();      // my wave's deposits for tile kt are in LDS
    __syncthreads();       // everyone's deposits visible; prior readers done

    if (kt + 1 < KT) {     // overlap next tile fetch with compute
      const int nxt = cur ^ 1;
      stage_tile<128>(Ablk + (kt + 1) * 32, &As[nxt][0], tid);
      stage_tile<256>(Bblk + (kt + 1) * 32, &Bs[nxt][0], tid);
    }

    const __bf16* a0 = &As[cur][0];
    const __bf16* b0 = &Bs[cur][0];

    v16bf a[4];
#pragma unroll
    for (int i = 0; i < 4; i++) {
      const __bf16* pa = a0 + (wm * 64 + i * 16 + r) * LDS_STRIDE + h * 8;
      v8bf lo = *(const v8bf*)pa;
      v8bf hi = *(const v8bf*)(pa + 16);
      a[i] = __builtin_shufflevector(lo, hi, 0, 1, 2, 3, 4, 5, 6, 7,
                                     8, 9, 10, 11, 12, 13, 14, 15);
    }
    v16bf b[4];
#pragma unroll
    for (int j = 0; j < 4; j++) {
      const __bf16* pb = b0 + (wn * 64 + j * 16 + r) * LDS_STRIDE + h * 8;
      v8bf lo = *(const v8bf*)pb;
      v8bf hi = *(const v8bf*)(pb + 16);
      b[j] = __builtin_shufflevector(lo, hi, 0, 1, 2, 3, 4, 5, 6, 7,
                                     8, 9, 10, 11, 12, 13, 14, 15);
    }
#pragma unroll
    for (int i = 0; i < 4; i++)
#pragma unroll
      for (int j = 0; j < 4; j++)
        acc[i][j] = __builtin_amdgcn_wmma_f32_16x16x32_bf16(
            false, a[i], false, b[j], (short)0, acc[i][j], false, false);
  }

  // Epilogue: D layout — lane n = n0+j*16+r, VGPR v -> row m0+i*16+h*8+v
  float lmin = 3.4e38f, lmax = -3.4e38f;
#pragma unroll
  for (int i = 0; i < 4; i++) {
#pragma unroll
    for (int j = 0; j < 4; j++) {
      const int n = n0 + j * 16 + r;
      const float bv = bias[n];
#pragma unroll
      for (int v = 0; v < 8; v++) {
        const int m = m0 + i * 16 + h * 8 + v;
        const float val = acc[i][j][v] + bv;
        out[(size_t)m * NDIM + n] = val;
        lmin = fminf(lmin, val);
        lmax = fmaxf(lmax, val);
      }
    }
  }

  __shared__ float smin[256], smax[256];
  smin[tid] = lmin;
  smax[tid] = lmax;
  __syncthreads();
  for (int o = 128; o > 0; o >>= 1) {
    if (tid < o) {
      smin[tid] = fminf(smin[tid], smin[tid + o]);
      smax[tid] = fmaxf(smax[tid], smax[tid + o]);
    }
    __syncthreads();
  }
  if (tid == 0) {
    const int bid = blockIdx.y * gridDim.x + blockIdx.x;
    bmin[bid] = smin[0];
    bmax[bid] = smax[0];
  }
}

// ---------------- 6) global min/max -> s = 255/(max-min) ----------------
__global__ __launch_bounds__(256) void minmax_final(const float* __restrict__ bmin,
                                                    const float* __restrict__ bmax,
                                                    float* __restrict__ sOut) {
  __shared__ float smin[256], smax[256];
  float mn = 3.4e38f, mx = -3.4e38f;
  for (int i = threadIdx.x; i < GEMM_BLOCKS; i += 256) {
    mn = fminf(mn, bmin[i]);
    mx = fmaxf(mx, bmax[i]);
  }
  smin[threadIdx.x] = mn;
  smax[threadIdx.x] = mx;
  __syncthreads();
  for (int o = 128; o > 0; o >>= 1) {
    if (threadIdx.x < (unsigned)o) {
      smin[threadIdx.x] = fminf(smin[threadIdx.x], smin[threadIdx.x + o]);
      smax[threadIdx.x] = fmaxf(smax[threadIdx.x], smax[threadIdx.x + o]);
    }
    __syncthreads();
  }
  if (threadIdx.x == 0) sOut[0] = 255.0f / (smax[0] - smin[0]);
}

// ---------------- 7) activation quantization: rint(v*s)/s (RNE == jnp.round) --
__global__ __launch_bounds__(256) void quant_act(float4* __restrict__ out,
                                                 const float* __restrict__ sPtr) {
  const float s = sPtr[0];
  const float inv = 1.0f / s;
  for (long long i = (long long)blockIdx.x * 256 + threadIdx.x; i < XELEMS / 4;
       i += (long long)gridDim.x * 256) {
    float4 v = out[i];
    v.x = rintf(v.x * s) * inv;
    v.y = rintf(v.y * s) * inv;
    v.z = rintf(v.z * s) * inv;
    v.w = rintf(v.w * s) * inv;
    out[i] = v;
  }
}

extern "C" void kernel_launch(void* const* d_in, const int* in_sizes, int n_in,
                              void* d_out, int out_size, void* d_ws, size_t ws_size,
                              hipStream_t stream) {
  const float* x      = (const float*)d_in[0];
  const float* weight = (const float*)d_in[1];
  const float* bias   = (const float*)d_in[2];
  const float* scale  = (const float*)d_in[3];
  float* out = (float*)d_out;

  // Workspace layout
  char* ws = (char*)d_ws;
  float* part = (float*)ws;                 // 1024 floats
  float* thr  = part + ABS_BLOCKS;          // 1
  float* sPtr = thr + 1;                    // 1
  float* bmin = (float*)(ws + 8192);        // 1024
  float* bmax = bmin + 2048;                // 1024 (capacity 2048)
  __bf16* Wq  = (__bf16*)(ws + 65536);      // 32 MiB
  __bf16* Xb  = Wq + WELEMS;                // 64 MiB

  abssum_partial<<<ABS_BLOCKS, 256, 0, stream>>>((const float4*)weight, part);
  finalize_thr<<<1, 256, 0, stream>>>(part, scale, thr);
  quant_w<<<1024, 256, 0, stream>>>((const float4*)weight, scale, thr, (v4bf*)Wq);
  cvt_x<<<2048, 256, 0, stream>>>((const float4*)x, (v4bf*)Xb);

  dim3 grid(NDIM / 256, MDIM / 128);  // 16 x 64
  gemm_bitlinear<<<grid, 256, 0, stream>>>(Xb, Wq, bias, out, bmin, bmax);

  minmax_final<<<1, 256, 0, stream>>>(bmin, bmax, sPtr);
  quant_act<<<2048, 256, 0, stream>>>((float4*)out, sPtr);
}
